// myBadTransfomerBlock_46617575031493
// MI455X (gfx1250) — matrix operand order repaired and checked by reading, not compiled, not run on hardware
//
#include <hip/hip_runtime.h>
#include <hip/hip_bf16.h>
#include <stdint.h>

// -------- fixed problem dims (from reference) --------
#define BATCH 16
#define SEQ   2048
#define DIM   1024

typedef __attribute__((ext_vector_type(16))) __bf16 v16bf;
typedef __attribute__((ext_vector_type(8)))  __bf16 bf16x8;
typedef __attribute__((ext_vector_type(4)))  __bf16 bf16x4;
typedef __attribute__((ext_vector_type(8)))  float  v8f;
typedef __attribute__((ext_vector_type(4)))  float  f32x4;

// -------- CDNA5 async global->LDS via inline asm (ISA: GLOBAL_LOAD_ASYNC_TO_LDS_B128)
static __device__ __forceinline__ void async_load_b128(const void* gptr, void* ldsptr) {
  unsigned lds32 = (unsigned)(uintptr_t)ldsptr;   // flat->LDS: addr[31:0]
  asm volatile("global_load_async_to_lds_b128 %0, %1, off"
               :
               : "v"(lds32), "v"(gptr)
               : "memory");
}
static __device__ __forceinline__ void wait_async_copies() {
#if __has_builtin(__builtin_amdgcn_s_wait_asynccnt)
  __builtin_amdgcn_s_wait_asynccnt(0);
#else
  asm volatile("s_wait_asynccnt 0x0" ::: "memory");
#endif
}

// -------- GEMM tiling --------
#define BM 128
#define BN 128
#define BK 64
#define LDS_K 72                 // BK + 8 bf16 pad (row stride 144B, 16B aligned)
// copy: BK/8 = 8 x b128 chunks per row; 256 threads -> 32 rows/iter -> 4 iters

static __device__ __forceinline__ void copy_tile(const __bf16* __restrict__ g, long ld,
                                                 __bf16* s, int tid) {
#pragma unroll
  for (int i = 0; i < 4; ++i) {
    int id  = i * 256 + tid;      // 0..1023
    int row = id >> 3;            // 128 rows
    int c8  = (id & 7) * 8;       // 0..56 step 8
    const __bf16* gp = g + (long)row * ld + c8;
    __bf16*       sp = s + row * LDS_K + c8;
    async_load_b128((const void*)gp, (void*)sp);
  }
}

static __device__ __forceinline__ v16bf combine16(bf16x8 lo, bf16x8 hi) {
  v16bf r;
#pragma unroll
  for (int i = 0; i < 8; ++i) { r[i] = lo[i]; r[i + 8] = hi[i]; }
  return r;
}

// A fragment 16x32 bf16: lanes 0-15 hold K {0..7,16..23}, lanes 16-31 hold K {8..15,24..31}
static __device__ __forceinline__ v16bf load_a_frag(const __bf16* s, int m16, int ks, int lane) {
  int m = m16 + (lane & 15);
  int kh = (lane >> 4) * 8;
  const __bf16* p = s + m * LDS_K + ks + kh;
  return combine16(*(const bf16x8*)p, *(const bf16x8*)(p + 16));
}
// B fragment 32x16 bf16 (NT: row n holds K run): lanes 0-15 K {0..7,16..23}, lanes 16-31 K {8..15,24..31}
static __device__ __forceinline__ v16bf load_b_frag(const __bf16* s, int n16, int ks, int lane) {
  int n = n16 + (lane & 15);
  int ko = (lane >> 4) * 8;
  const __bf16* p = s + n * LDS_K + ks + ko;
  return combine16(*(const bf16x8*)p, *(const bf16x8*)(p + 16));
}

// C[m,n] = sum_k A[m,k] * Bt[n,k]; A: MxK bf16 row-major, Bt: NxK bf16 row-major.
template <int RELU, int OUTF32>
__global__ __launch_bounds__(256) void gemm_nt_bf16(
    const __bf16* __restrict__ A, const __bf16* __restrict__ Bt, void* __restrict__ Cv,
    int M, int N, int K, long strideA, long strideB, long strideC) {
  __shared__ __bf16 sA[2][BM * LDS_K];
  __shared__ __bf16 sB[2][BN * LDS_K];

  const int tid  = threadIdx.x;
  const int lane = tid & 31;
  const int wave = tid >> 5;
  const int z    = blockIdx.z;
  A  += (long)z * strideA;
  Bt += (long)z * strideB;

  const int m0 = blockIdx.x * BM;
  const int n0 = blockIdx.y * BN;
  const int wm = (wave & 3) * 32;   // 4 waves along M
  const int wn = (wave >> 2) * 64;  // 2 waves along N

  v8f acc[2][4] = {};

  const int nk = K / BK;
  copy_tile(A + (long)m0 * K, K, sA[0], tid);
  copy_tile(Bt + (long)n0 * K, K, sB[0], tid);
  wait_async_copies();
  __syncthreads();

  for (int kt = 0; kt < nk; ++kt) {
    const int cur = kt & 1, nxt = cur ^ 1;
    if (kt + 1 < nk) {
      copy_tile(A + (long)m0 * K + (long)(kt + 1) * BK, K, sA[nxt], tid);
      copy_tile(Bt + (long)n0 * K + (long)(kt + 1) * BK, K, sB[nxt], tid);
    }
#pragma unroll
    for (int ks = 0; ks < BK; ks += 32) {
      v16bf af0 = load_a_frag(sA[cur], wm, ks, lane);
      v16bf af1 = load_a_frag(sA[cur], wm + 16, ks, lane);
      v16bf bf0 = load_b_frag(sB[cur], wn, ks, lane);
      v16bf bf1 = load_b_frag(sB[cur], wn + 16, ks, lane);
      v16bf bf2 = load_b_frag(sB[cur], wn + 32, ks, lane);
      v16bf bf3 = load_b_frag(sB[cur], wn + 48, ks, lane);

      acc[0][0] = __builtin_amdgcn_wmma_f32_16x16x32_bf16(false, af0, false, bf0, (short)0, acc[0][0], false, false);
      acc[0][1] = __builtin_amdgcn_wmma_f32_16x16x32_bf16(false, af0, false, bf1, (short)0, acc[0][1], false, false);
      acc[0][2] = __builtin_amdgcn_wmma_f32_16x16x32_bf16(false, af0, false, bf2, (short)0, acc[0][2], false, false);
      acc[0][3] = __builtin_amdgcn_wmma_f32_16x16x32_bf16(false, af0, false, bf3, (short)0, acc[0][3], false, false);
      acc[1][0] = __builtin_amdgcn_wmma_f32_16x16x32_bf16(false, af1, false, bf0, (short)0, acc[1][0], false, false);
      acc[1][1] = __builtin_amdgcn_wmma_f32_16x16x32_bf16(false, af1, false, bf1, (short)0, acc[1][1], false, false);
      acc[1][2] = __builtin_amdgcn_wmma_f32_16x16x32_bf16(false, af1, false, bf2, (short)0, acc[1][2], false, false);
      acc[1][3] = __builtin_amdgcn_wmma_f32_16x16x32_bf16(false, af1, false, bf3, (short)0, acc[1][3], false, false);
    }
    wait_async_copies();
    __syncthreads();
  }

  // epilogue: C/D layout — VGPR r, lanes 0-15: M=r, N=lane; lanes 16-31: M=8+r, N=lane-16
  float*  Cf = (float*)Cv  + (long)z * strideC;
  __bf16* Cb = (__bf16*)Cv + (long)z * strideC;
#pragma unroll
  for (int i = 0; i < 2; ++i) {
#pragma unroll
    for (int j = 0; j < 4; ++j) {
      const int mbase = m0 + wm + i * 16 + ((lane >> 4) * 8);
      const int n     = n0 + wn + j * 16 + (lane & 15);
#pragma unroll
      for (int r = 0; r < 8; ++r) {
        float v = acc[i][j][r];
        if (RELU) v = fmaxf(v, 0.0f);
        long off = (long)(mbase + r) * N + n;
        if (OUTF32) Cf[off] = v; else Cb[off] = (__bf16)v;
      }
    }
  }
}

// -------- elementwise kernels --------
__global__ __launch_bounds__(256) void k_cvt_w(const float* __restrict__ w,
                                               __bf16* __restrict__ o, long n4) {
  long i = (long)blockIdx.x * 256 + threadIdx.x;
  if (i >= n4) return;
  f32x4 v = *(const f32x4*)(w + i * 4);
  bf16x4 b;
#pragma unroll
  for (int k = 0; k < 4; ++k) b[k] = (__bf16)v[k];
  *(bf16x4*)(o + i * 4) = b;
}

__global__ __launch_bounds__(256) void k_addpos(const float* __restrict__ x,
                                                const float* __restrict__ pos,
                                                __bf16* __restrict__ y, long sd, long n4) {
  long i = (long)blockIdx.x * 256 + threadIdx.x;
  if (i >= n4) return;
  long e = i * 4;
  f32x4 xv = *(const f32x4*)(x + e);
  f32x4 pv = *(const f32x4*)(pos + (e % sd));
  bf16x4 b;
#pragma unroll
  for (int k = 0; k < 4; ++k) b[k] = (__bf16)(xv[k] + pv[k]);
  *(bf16x4*)(y + e) = b;
}

// per-row L2 normalize; writes y [B,S,D] and yT [B,D,S] (both bf16)
__global__ __launch_bounds__(256) void k_norm(const __bf16* __restrict__ h,
                                              __bf16* __restrict__ y,
                                              __bf16* __restrict__ yT) {
  const long row = blockIdx.x;              // b*SEQ + s
  const long b   = row / SEQ;
  const long s   = row % SEQ;
  const int  tid = threadIdx.x;
  bf16x4 hv = *(const bf16x4*)(h + row * DIM + tid * 4);
  float v[4]; float ss = 0.0f;
#pragma unroll
  for (int k = 0; k < 4; ++k) { v[k] = (float)hv[k]; ss += v[k] * v[k]; }
#pragma unroll
  for (int o = 16; o > 0; o >>= 1) ss += __shfl_xor(ss, o, 32);
  __shared__ float wsum[8];
  if ((tid & 31) == 0) wsum[tid >> 5] = ss;
  __syncthreads();
  float tot = 0.0f;
#pragma unroll
  for (int w = 0; w < 8; ++w) tot += wsum[w];
  const float inv = 1.0f / (sqrtf(tot) + 1e-6f);
  bf16x4 ov;
#pragma unroll
  for (int k = 0; k < 4; ++k) ov[k] = (__bf16)(v[k] * inv);
  *(bf16x4*)(y + row * DIM + tid * 4) = ov;
#pragma unroll
  for (int k = 0; k < 4; ++k) {
    long d = tid * 4 + k;
    yT[(b * DIM + d) * (long)SEQ + s] = ov[k];
  }
}

// -------- host launch --------
extern "C" void kernel_launch(void* const* d_in, const int* in_sizes, int n_in,
                              void* d_out, int out_size, void* d_ws, size_t ws_size,
                              hipStream_t stream) {
  const float* x    = (const float*)d_in[0];
  const float* pos  = (const float*)d_in[1];
  const float* ew0  = (const float*)d_in[2];
  const float* ew1  = (const float*)d_in[3];
  const float* dw0  = (const float*)d_in[4];
  const float* dw1  = (const float*)d_in[5];

  const long MTOT = (long)BATCH * SEQ;          // 32768 token rows
  const long ACT  = MTOT * DIM;                 // 33,554,432 elems
  const size_t MB = 1ull << 20;

  char* ws = (char*)d_ws;
  __bf16* b0  = (__bf16*)(ws + 0 * MB);         // 64 MB
  __bf16* b1  = (__bf16*)(ws + 64 * MB);        // 64 MB
  __bf16* b2  = (__bf16*)(ws + 128 * MB);       // 64 MB (yT)
  __bf16* G   = (__bf16*)(ws + 192 * MB);       // 32 MB (Gram, per-batch DxD)
  __bf16* w0b = (__bf16*)(ws + 224 * MB);
  __bf16* w1b = (__bf16*)(ws + 226 * MB);
  __bf16* w2b = (__bf16*)(ws + 228 * MB);
  __bf16* w3b = (__bf16*)(ws + 230 * MB);       // total 232 MB

  // weights fp32 -> bf16
  const long w4 = (long)DIM * DIM / 4;
  k_cvt_w<<<(w4 + 255) / 256, 256, 0, stream>>>(ew0, w0b, w4);
  k_cvt_w<<<(w4 + 255) / 256, 256, 0, stream>>>(ew1, w1b, w4);
  k_cvt_w<<<(w4 + 255) / 256, 256, 0, stream>>>(dw0, w2b, w4);
  k_cvt_w<<<(w4 + 255) / 256, 256, 0, stream>>>(dw1, w3b, w4);

  // pos-embed add -> b0 (bf16)
  const long a4 = ACT / 4;
  k_addpos<<<(a4 + 255) / 256, 256, 0, stream>>>(x, pos, b0, (long)SEQ * DIM, a4);

  dim3 blk(256);
  dim3 g_mlp(MTOT / BM, DIM / BN, 1);

  // encoder MLP
  gemm_nt_bf16<1, 0><<<g_mlp, blk, 0, stream>>>(b0, w0b, b1, (int)MTOT, DIM, DIM, 0, 0, 0);
  gemm_nt_bf16<1, 0><<<g_mlp, blk, 0, stream>>>(b1, w1b, b0, (int)MTOT, DIM, DIM, 0, 0, 0);

  // L2 normalize -> y (b1) and yT (b2)
  k_norm<<<(int)MTOT, blk, 0, stream>>>(b0, b1, b2);

  // Gram: G[b] = Y^T Y  (A=Bt=yT, per-batch DxS)  — (YY^T)Y == Y(Y^T Y)
  dim3 g_gram(DIM / BM, DIM / BN, BATCH);
  gemm_nt_bf16<0, 0><<<g_gram, blk, 0, stream>>>(b2, b2, G, DIM, DIM, SEQ,
                                                 (long)DIM * SEQ, (long)DIM * SEQ,
                                                 (long)DIM * DIM);
  // attention out: A = Y (SxD), Bt = G (symmetric) -> b0
  dim3 g_att(SEQ / BM, DIM / BN, BATCH);
  gemm_nt_bf16<0, 0><<<g_att, blk, 0, stream>>>(b1, G, b0, SEQ, DIM, DIM,
                                                (long)SEQ * DIM, (long)DIM * DIM,
                                                (long)SEQ * DIM);

  // decoder MLP; final layer writes fp32 d_out
  gemm_nt_bf16<1, 0><<<g_mlp, blk, 0, stream>>>(b0, w2b, b1, (int)MTOT, DIM, DIM, 0, 0, 0);
  gemm_nt_bf16<1, 1><<<g_mlp, blk, 0, stream>>>(b1, w3b, d_out, (int)MTOT, DIM, DIM, 0, 0, 0);

  (void)in_sizes; (void)n_in; (void)out_size; (void)ws_size;
}
